// MultiScenarioCloneBase_65781719106228
// MI455X (gfx1250) — matrix operand order repaired and checked by reading, not compiled, not run on hardware
//
#include <hip/hip_runtime.h>
#include <stdint.h>

#define BATCH 65536
#define FDIM  128
#define HDIM  256
#define NDOM  8
#define TM    64        // samples per workgroup tile
#define AST   68        // padded LDS row stride in u32 (64 data words + 4 pad)

// pre-swizzled W1 fragments: frag = ((dom*16 + ntile)*4 + kb), 512 bf16 per frag
#define NFRAG       (NDOM * 16 * 4)
#define W1SWZ_ELEMS (NFRAG * 512)     // 262144 per (hi|lo) array

typedef __bf16 bf16_t;
typedef bf16_t v16bf __attribute__((ext_vector_type(16)));
typedef float  v8f   __attribute__((ext_vector_type(8)));

struct U32x8 { uint4 a, b; };

static __device__ __forceinline__ unsigned short bfbits(bf16_t b) {
    return __builtin_bit_cast(unsigned short, b);
}
// pack two bf16 (col 2j -> low half, col 2j+1 -> high half; little-endian order)
static __device__ __forceinline__ unsigned pack2(bf16_t e0, bf16_t e1) {
    return (unsigned)bfbits(e0) | ((unsigned)bfbits(e1) << 16);
}

// ---------------- bucketing kernels ----------------
// ws layout (u32): [0..7]=counts, [8..15]=offsets, [16..23]=cursors,
//                  [32..32+B)=perm, then bf16 Whi[262144], Wlo[262144]

__global__ void k_zero(unsigned* ws) {
    if (threadIdx.x < NDOM) ws[threadIdx.x] = 0u;
}

__global__ void k_count(const int* __restrict__ ids, unsigned* __restrict__ ws) {
    int b = blockIdx.x * blockDim.x + threadIdx.x;
    if (b < BATCH) atomicAdd(&ws[ids[b]], 1u);
}

__global__ void k_scan(unsigned* ws) {
    if (threadIdx.x == 0) {
        unsigned off = 0;
        for (int d = 0; d < NDOM; ++d) {
            ws[8 + d]  = off;
            ws[16 + d] = off;
            off += ws[d];
        }
    }
}

__global__ void k_scatter(const int* __restrict__ ids, unsigned* __restrict__ ws,
                          unsigned* __restrict__ perm) {
    int b = blockIdx.x * blockDim.x + threadIdx.x;
    if (b < BATCH) {
        unsigned p = atomicAdd(&ws[16 + ids[b]], 1u);
        perm[p] = (unsigned)b;
    }
}

// ---------------- one-time W1 split + swizzle into contiguous-K B-fragment layout ----
// B fragment as stored here (32 x 16, contiguous K): lane half 0 holds K=0..15, half 1
// holds K=16..31; n = lane&15. Element order in memory: lane*16 + i.
__global__ void k_splitw1(const float* __restrict__ W1, bf16_t* __restrict__ whi,
                          bf16_t* __restrict__ wlo) {
    int e = blockIdx.x * blockDim.x + threadIdx.x;
    if (e >= W1SWZ_ELEMS) return;
    int idx   = e & 511;
    int frag  = e >> 9;
    int kb    = frag & 3;
    int ntile = (frag >> 2) & 15;
    int dom   = frag >> 6;
    int lane  = idx >> 4;
    int i     = idx & 15;
    int f = kb * 32 + i + ((lane >> 4) << 4);
    int n = ntile * 16 + (lane & 15);
    float v = W1[((size_t)dom * FDIM + f) * HDIM + n];
    bf16_t h = (bf16_t)v;
    whi[e] = h;
    wlo[e] = (bf16_t)(v - (float)h);
}

// ---------------- fused tower kernel (WMMA) ----------------
// Block = 256 threads (8 wave32). One block handles 64 samples of one domain:
//   h = relu(x @ W1[d] + b1[d]);  out = h @ W2[d] + b2[d]  (OUT=1)
// Layer-1 GEMM: M=64 x N=256 x K=128 via v_wmma_f32_16x16x32_bf16 with bf16
// hi/lo split (3 WMMAs per tile step) for ~fp32 accuracy.

__global__ __launch_bounds__(256, 1) void k_towers(
    const float* __restrict__ x,
    const bf16_t* __restrict__ whi, const bf16_t* __restrict__ wlo,
    const float* __restrict__ b1, const float* __restrict__ W2,
    const float* __restrict__ b2, const unsigned* __restrict__ counts,
    const unsigned* __restrict__ offsets, const unsigned* __restrict__ perm,
    float* __restrict__ out)
{
    __shared__ unsigned Ahi[TM * AST];   // bf16 hi pairs: word j = cols {2j, 2j+1}
    __shared__ unsigned Alo[TM * AST];   // bf16 lo pairs
    __shared__ float    partial[8][TM];  // per-wave layer-2 partials

    const int dom = blockIdx.y;
    const unsigned cnt_d = counts[dom];
    const unsigned tile  = blockIdx.x;
    if (tile * TM >= cnt_d) return;      // uniform early-exit (before barriers)
    const unsigned base = offsets[dom] + tile * TM;
    const int cnt = (int)((cnt_d - tile * TM) < (unsigned)TM ? (cnt_d - tile * TM)
                                                             : (unsigned)TM);

    const int t = threadIdx.x;

    // ---- stage gathered x rows -> LDS split bf16 hi/lo (zeros for pad rows) ----
    {
        const int p    = t & 63;             // column pair: cols 2p, 2p+1
        const int rsel = t >> 6;             // 0..3
        for (int it = 0; it < TM / 4; ++it) {
            int row = it * 4 + rsel;
            float v0 = 0.f, v1 = 0.f;
            if (row < cnt) {
                const float* xr = x + (size_t)perm[base + row] * FDIM + 2 * p;
                v0 = xr[0];
                v1 = xr[1];
            }
            bf16_t h0 = (bf16_t)v0, h1 = (bf16_t)v1;
            bf16_t l0 = (bf16_t)(v0 - (float)h0), l1 = (bf16_t)(v1 - (float)h1);
            Ahi[row * AST + p] = pack2(h0, h1);
            Alo[row * AST + p] = pack2(l0, l1);
        }
    }
    __syncthreads();

    const int w    = t >> 5;             // wave 0..7 -> owns H columns [w*32, w*32+32)
    const int lane = t & 31;
    const int l15  = lane & 15;
    const int lhi  = lane >> 4;          // lane half (ISA 16-lane groups)

    const float* b1d = b1 + dom * HDIM;
    const float* W2d = W2 + dom * HDIM;  // OUT == 1

    float b1v[2], w2v[2];
    size_t fragoff[2];                   // bf16 offset of this lane's fragment base
#pragma unroll
    for (int nt = 0; nt < 2; ++nt) {
        int n = w * 32 + nt * 16 + l15;
        b1v[nt] = b1d[n];
        w2v[nt] = W2d[n];
        fragoff[nt] = (size_t)((dom * 16 + w * 2 + nt) * 4) * 512 + lane * 16;
    }

    // Accumulators C[m-tile][n-tile]; bias b1 folded into the init (all 8 rows in a
    // lane share column n = l15).
    v8f C[4][2];
#pragma unroll
    for (int mt = 0; mt < 4; ++mt)
#pragma unroll
        for (int nt = 0; nt < 2; ++nt)
#pragma unroll
            for (int r = 0; r < 8; ++r)
                C[mt][nt][r] = b1v[nt];

    for (int kb = 0; kb < 4; ++kb) {
        // ---- B fragments: single aligned 32B vector load per lane (pre-swizzled)
        v16bf bh[2], bl[2];
#pragma unroll
        for (int nt = 0; nt < 2; ++nt) {
            size_t off = fragoff[nt] + (size_t)kb * 512;
            bh[nt] = *(const v16bf*)(whi + off);
            bl[nt] = *(const v16bf*)(wlo + off);
        }
        // ---- A fragments straight from LDS (contiguous K order, matching B):
        //      elements 0-15 = K k0..k0+15, k0 = kb*32+lhi*16
        const int wbase = (kb * 32 + (lhi << 4)) >> 1;   // u32 word index (mult of 4)
#pragma unroll
        for (int mt = 0; mt < 4; ++mt) {
            const int rb = (mt * 16 + l15) * AST;
            U32x8 th, tl;
            th.a = *(const uint4*)&Ahi[rb + wbase];
            th.b = *(const uint4*)&Ahi[rb + wbase + 4];
            tl.a = *(const uint4*)&Alo[rb + wbase];
            tl.b = *(const uint4*)&Alo[rb + wbase + 4];
            v16bf ahi = __builtin_bit_cast(v16bf, th);
            v16bf alo = __builtin_bit_cast(v16bf, tl);
#pragma unroll
            for (int nt = 0; nt < 2; ++nt) {
                C[mt][nt] = __builtin_amdgcn_wmma_f32_16x16x32_bf16(
                    false, ahi, false, bh[nt], (short)0, C[mt][nt], false, false);
                C[mt][nt] = __builtin_amdgcn_wmma_f32_16x16x32_bf16(
                    false, ahi, false, bl[nt], (short)0, C[mt][nt], false, false);
                C[mt][nt] = __builtin_amdgcn_wmma_f32_16x16x32_bf16(
                    false, alo, false, bh[nt], (short)0, C[mt][nt], false, false);
            }
        }
    }

    // ---- fused layer 2: relu, * W2, reduce over this wave's 32 columns ----
#pragma unroll
    for (int mt = 0; mt < 4; ++mt) {
        float lp[8];
#pragma unroll
        for (int r = 0; r < 8; ++r) lp[r] = 0.f;
#pragma unroll
        for (int nt = 0; nt < 2; ++nt)
#pragma unroll
            for (int r = 0; r < 8; ++r) {
                float hv = C[mt][nt][r];
                hv = hv > 0.f ? hv : 0.f;
                lp[r] += hv * w2v[nt];
            }
        // reduce across the 16 lanes of each half (xor masks < 16 stay in-half)
#pragma unroll
        for (int r = 0; r < 8; ++r) {
            lp[r] += __shfl_xor(lp[r], 1, 32);
            lp[r] += __shfl_xor(lp[r], 2, 32);
            lp[r] += __shfl_xor(lp[r], 4, 32);
            lp[r] += __shfl_xor(lp[r], 8, 32);
        }
        if (l15 == 0) {   // lane 0 -> rows mt*16+r ; lane 16 -> rows mt*16+8+r
#pragma unroll
            for (int r = 0; r < 8; ++r)
                partial[w][mt * 16 + (lhi << 3) + r] = lp[r];
        }
    }
    __syncthreads();

    // fixed-order cross-wave combine (deterministic), + b2, scatter store
    if (t < TM && t < cnt) {
        float s = b2[dom];
#pragma unroll
        for (int ww = 0; ww < 8; ++ww) s += partial[ww][t];
        out[perm[base + t]] = s;
    }
}

// ---------------- launcher ----------------
extern "C" void kernel_launch(void* const* d_in, const int* in_sizes, int n_in,
                              void* d_out, int out_size, void* d_ws, size_t ws_size,
                              hipStream_t stream) {
    (void)in_sizes; (void)n_in; (void)out_size; (void)ws_size;
    const float* x  = (const float*)d_in[0];
    const int*   ids= (const int*)  d_in[1];
    const float* W1 = (const float*)d_in[2];
    const float* b1 = (const float*)d_in[3];
    const float* W2 = (const float*)d_in[4];
    const float* b2 = (const float*)d_in[5];
    float* out = (float*)d_out;

    unsigned* ws   = (unsigned*)d_ws;
    unsigned* perm = ws + 32;                      // [32 .. 32+BATCH)
    bf16_t*   whi  = (bf16_t*)(ws + 32 + BATCH);   // 262144 bf16 (512 KB)
    bf16_t*   wlo  = whi + W1SWZ_ELEMS;            // 262144 bf16 (512 KB)

    hipLaunchKernelGGL(k_zero,    dim3(1),           dim3(32),  0, stream, ws);
    hipLaunchKernelGGL(k_count,   dim3(BATCH / 256), dim3(256), 0, stream, ids, ws);
    hipLaunchKernelGGL(k_scan,    dim3(1),           dim3(32),  0, stream, ws);
    hipLaunchKernelGGL(k_scatter, dim3(BATCH / 256), dim3(256), 0, stream, ids, ws, perm);
    hipLaunchKernelGGL(k_splitw1, dim3(W1SWZ_ELEMS / 256), dim3(256), 0, stream,
                       W1, whi, wlo);
    hipLaunchKernelGGL(k_towers,  dim3(BATCH / TM, NDOM), dim3(256), 0, stream,
                       x, whi, wlo, b1, W2, b2, ws, ws + 8, perm, out);
}
